// SpKBGATModified_50414326120605
// MI455X (gfx1250) — matrix operand order repaired and checked by reading, not compiled, not run on hardware
//
#include <hip/hip_runtime.h>
#include <hip/hip_bf16.h>
#include <math.h>

typedef __attribute__((ext_vector_type(16))) _Float16 v16h;
typedef __attribute__((ext_vector_type(8)))  _Float16 v8h;
typedef __attribute__((ext_vector_type(8)))  float    v8f;

#define N_NODES   50000
#define N_REL     256
#define D1        128
#define D2        256
#define E1_CNT    300000
#define E2_CNT    30000
#define ET_CNT    330000
#define LRELU_A   0.2f

static __device__ __forceinline__ float eluf(float v) {
  return v > 0.f ? v : (__expf(v) - 1.f);
}
static __device__ __forceinline__ float wave_sum32(float v) {
  #pragma unroll
  for (int off = 16; off > 0; off >>= 1) v += __shfl_xor(v, off, 32);
  return v;
}

// ---------------- utility kernels ----------------
__global__ void zero_f32(float* p, size_t n) {
  size_t i = (size_t)blockIdx.x * blockDim.x + threadIdx.x;
  size_t stride = (size_t)gridDim.x * blockDim.x;
  for (; i < n; i += stride) p[i] = 0.f;
}

// l2-normalize entity rows (d=128), emit f32 and f16 copies. One wave per node.
__global__ void l2norm128_kernel(const float* __restrict__ in, float* __restrict__ out,
                                 _Float16* __restrict__ outh) {
  int n = blockIdx.x * (blockDim.x >> 5) + (threadIdx.x >> 5);
  if (n >= N_NODES) return;
  int lane = threadIdx.x & 31;
  const float4 v = *(const float4*)(in + (size_t)n * D1 + lane * 4);
  float s = v.x * v.x + v.y * v.y + v.z * v.z + v.w * v.w;
  s = wave_sum32(s);
  float inv = 1.f / fmaxf(sqrtf(s), 1e-12f);
  size_t base = (size_t)n * D1 + lane * 4;
  out[base + 0] = v.x * inv; out[base + 1] = v.y * inv;
  out[base + 2] = v.z * inv; out[base + 3] = v.w * inv;
  outh[base + 0] = (_Float16)(v.x * inv); outh[base + 1] = (_Float16)(v.y * inv);
  outh[base + 2] = (_Float16)(v.z * inv); outh[base + 3] = (_Float16)(v.w * inv);
}

// pack f32 weight slice [rows, cols] (row stride ld, col offset off) -> tight f16 [rows,cols]
__global__ void pack_f16(const float* __restrict__ src, int ld, int off,
                         _Float16* __restrict__ dst, int rows, int cols) {
  int idx = blockIdx.x * blockDim.x + threadIdx.x;
  if (idx >= rows * cols) return;
  int r = idx / cols, c = idx - r * cols;
  dst[idx] = (_Float16)src[(size_t)r * ld + off + c];
}

// transpose-pack: src [K,N] f32 -> dst [N,K] f16
__global__ void pack_f16_t(const float* __restrict__ src, int K, int N,
                           _Float16* __restrict__ dst) {
  int idx = blockIdx.x * blockDim.x + threadIdx.x;
  if (idx >= N * K) return;
  int j = idx / K, k = idx - j * K;
  dst[idx] = (_Float16)src[(size_t)k * N + j];
}

// small f32 GEMM: C[M,N] = A[M,K] @ B[K,N]  (thread per element)
__global__ void gemm_nn_f32(const float* __restrict__ A, const float* __restrict__ B,
                            float* __restrict__ C, int M, int N, int K) {
  int idx = blockIdx.x * blockDim.x + threadIdx.x;
  if (idx >= M * N) return;
  int r = idx / N, j = idx - r * N;
  float s = 0.f;
  for (int k = 0; k < K; ++k) s += A[(size_t)r * K + k] * B[(size_t)k * N + j];
  C[idx] = s;
}

// small f32 GEMM: C[M,N] = A[M,K] @ Bt^T where Bt[j, off+k], row stride ldb
__global__ void gemm_nt_f32(const float* __restrict__ A, const float* __restrict__ Bt,
                            int ldb, int off, float* __restrict__ C, int M, int N, int K) {
  int idx = blockIdx.x * blockDim.x + threadIdx.x;
  if (idx >= M * N) return;
  int r = idx / N, j = idx - r * N;
  float s = 0.f;
  for (int k = 0; k < K; ++k) s += A[(size_t)r * K + k] * Bt[(size_t)j * ldb + off + k];
  C[idx] = s;
}

// q[row] = dot(P[row,:D], vec)  — one wave per row
__global__ void rowdot(const float* __restrict__ P, const float* __restrict__ vec,
                       float* __restrict__ q, int D, int rows) {
  int r = blockIdx.x * (blockDim.x >> 5) + (threadIdx.x >> 5);
  if (r >= rows) return;
  int lane = threadIdx.x & 31;
  float s = 0.f;
  for (int c = lane; c < D; c += 32) s += P[(size_t)r * D + c] * vec[c];
  s = wave_sum32(s);
  if (lane == 0) q[r] = s;
}

// ---------------- WMMA GEMM ----------------
// C[M,N] f32 = A[M,K] f16 @ Bt[N,K]^T f16.  lda=K, ldb = Bt row stride (elems), ldc given.
// One wave computes a 16(M) x 64(N) tile; A fragment reused across 4 column subtiles.
// Fragment layouts (wave32, 16-bit A 16x32 / B 32x16 / f32 C 16x16): A and B share contiguous k = 16*kh + s.
__global__ void wmma_gemm_nt(const _Float16* __restrict__ A, int lda,
                             const _Float16* __restrict__ Bt, int ldb,
                             float* __restrict__ C, int ldc,
                             int M, int N, int K) {
  const int wave = blockIdx.x * (blockDim.x >> 5) + (threadIdx.x >> 5);
  const int lane = threadIdx.x & 31;
  const int ntiles = N >> 6;
  const int mt = wave / ntiles;
  const int nt = wave - mt * ntiles;
  if (mt * 16 >= M) return;
  const int l15 = lane & 15;
  const int kh  = lane >> 4;
  const int n0  = nt * 64;
  const _Float16* arow = A + (size_t)(mt * 16 + l15) * lda;
  const _Float16* bbase = Bt + (size_t)(n0 + l15) * ldb;
  v8f acc0 = {}, acc1 = {}, acc2 = {}, acc3 = {};
  for (int k0 = 0; k0 < K; k0 += 32) {
    // A fragment: 16 contiguous K values at k0 + kh*16 (matches B)
    v8h alo = *(const v8h*)(arow + k0 + kh * 16);
    v8h ahi = *(const v8h*)(arow + k0 + kh * 16 + 8);
    v16h a;
    #pragma unroll
    for (int i = 0; i < 8; ++i) { a[i] = alo[i]; a[8 + i] = ahi[i]; }
    // B fragments: 16 contiguous K values at k0 + kh*16 for column n0 + t*16 + l15
    const _Float16* bcol = bbase + k0 + kh * 16;
    v16h b0 = *(const v16h*)(bcol);
    v16h b1 = *(const v16h*)(bcol + (size_t)16 * ldb);
    v16h b2 = *(const v16h*)(bcol + (size_t)32 * ldb);
    v16h b3 = *(const v16h*)(bcol + (size_t)48 * ldb);
    acc0 = __builtin_amdgcn_wmma_f32_16x16x32_f16(false, a, false, b0, (short)0, acc0, false, false);
    acc1 = __builtin_amdgcn_wmma_f32_16x16x32_f16(false, a, false, b1, (short)0, acc1, false, false);
    acc2 = __builtin_amdgcn_wmma_f32_16x16x32_f16(false, a, false, b2, (short)0, acc2, false, false);
    acc3 = __builtin_amdgcn_wmma_f32_16x16x32_f16(false, a, false, b3, (short)0, acc3, false, false);
  }
  // D layout: M = v + 8*kh, N = l15
  float* crow = C + (size_t)(mt * 16 + 8 * kh) * ldc + n0 + l15;
  #pragma unroll
  for (int v = 0; v < 8; ++v) {
    crow[(size_t)v * ldc + 0]  = acc0[v];
    crow[(size_t)v * ldc + 16] = acc1[v];
    crow[(size_t)v * ldc + 32] = acc2[v];
    crow[(size_t)v * ldc + 48] = acc3[v];
  }
}

// ---------------- edge scatter kernels ----------------
// layer 1 (d=128): one wave per edge; 4 comps/lane
__global__ void gat_edge_d128(const int* __restrict__ el, const int* __restrict__ et,
                              const int* __restrict__ elnh, const int* __restrict__ etnh,
                              const float* __restrict__ pd, const float* __restrict__ ps,
                              const float* __restrict__ qd, const float* __restrict__ qs,
                              const float* __restrict__ pr, const float* __restrict__ qr,
                              float* __restrict__ hraw, float* __restrict__ rowsum) {
  int e = blockIdx.x * (blockDim.x >> 5) + (threadIdx.x >> 5);
  if (e >= ET_CNT) return;
  int lane = threadIdx.x & 31;
  int dst, src, r0, r1;
  if (e < E1_CNT) {
    dst = el[e]; src = el[E1_CNT + e]; r0 = et[e]; r1 = -1;
  } else {
    int en = e - E1_CNT;
    dst = elnh[en]; src = elnh[E2_CNT + en];
    r0 = etnh[2 * en]; r1 = etnh[2 * en + 1];
  }
  float q = qd[dst] + qs[src] + qr[r0] + (r1 >= 0 ? qr[r1] : 0.f);
  float p = (q > 0.f) ? -q : (-LRELU_A * q);   // -leaky_relu(q)
  float ee = __expf(p);
  if (lane == 0) atomicAdd(rowsum + dst, ee);
  int c = lane * 4;
  const float4 vd = *(const float4*)(pd + (size_t)dst * D1 + c);
  const float4 vs = *(const float4*)(ps + (size_t)src * D1 + c);
  float4 vr = *(const float4*)(pr + (size_t)r0 * D1 + c);
  if (r1 >= 0) {
    const float4 w = *(const float4*)(pr + (size_t)r1 * D1 + c);
    vr.x += w.x; vr.y += w.y; vr.z += w.z; vr.w += w.w;
  }
  float* o = hraw + (size_t)dst * D1 + c;
  atomicAdd(o + 0, ee * (vd.x + vs.x + vr.x));
  atomicAdd(o + 1, ee * (vd.y + vs.y + vr.y));
  atomicAdd(o + 2, ee * (vd.z + vs.z + vr.z));
  atomicAdd(o + 3, ee * (vd.w + vs.w + vr.w));
}

// layer 2 (d=256): one wave per edge; 8 comps/lane
__global__ void gat_edge_d256(const int* __restrict__ el, const int* __restrict__ et,
                              const int* __restrict__ elnh, const int* __restrict__ etnh,
                              const float* __restrict__ pd, const float* __restrict__ ps,
                              const float* __restrict__ qd, const float* __restrict__ qs,
                              const float* __restrict__ pr, const float* __restrict__ qr,
                              float* __restrict__ xlraw, float* __restrict__ rowsum) {
  int e = blockIdx.x * (blockDim.x >> 5) + (threadIdx.x >> 5);
  if (e >= ET_CNT) return;
  int lane = threadIdx.x & 31;
  int dst, src, r0, r1;
  if (e < E1_CNT) {
    dst = el[e]; src = el[E1_CNT + e]; r0 = et[e]; r1 = -1;
  } else {
    int en = e - E1_CNT;
    dst = elnh[en]; src = elnh[E2_CNT + en];
    r0 = etnh[2 * en]; r1 = etnh[2 * en + 1];
  }
  float q = qd[dst] + qs[src] + qr[r0] + (r1 >= 0 ? qr[r1] : 0.f);
  float p = (q > 0.f) ? -q : (-LRELU_A * q);
  float ee = __expf(p);
  if (lane == 0) atomicAdd(rowsum + dst, ee);
  int c = lane * 8;
  #pragma unroll
  for (int half = 0; half < 2; ++half) {
    int cc = c + half * 4;
    const float4 vd = *(const float4*)(pd + (size_t)dst * D2 + cc);
    const float4 vs = *(const float4*)(ps + (size_t)src * D2 + cc);
    float4 vr = *(const float4*)(pr + (size_t)r0 * D2 + cc);
    if (r1 >= 0) {
      const float4 w = *(const float4*)(pr + (size_t)r1 * D2 + cc);
      vr.x += w.x; vr.y += w.y; vr.z += w.z; vr.w += w.w;
    }
    float* o = xlraw + (size_t)dst * D2 + cc;
    atomicAdd(o + 0, ee * (vd.x + vs.x + vr.x));
    atomicAdd(o + 1, ee * (vd.y + vs.y + vr.y));
    atomicAdd(o + 2, ee * (vd.z + vs.z + vr.z));
    atomicAdd(o + 3, ee * (vd.w + vs.w + vr.w));
  }
}

// layer-1 node epilogue: h = elu(h_raw / rowsum) -> x[:, head*128 + c] (f32 + f16)
__global__ void l1_node(const float* __restrict__ hraw, const float* __restrict__ rowsum,
                        float* __restrict__ x, _Float16* __restrict__ xh, int head) {
  int idx = blockIdx.x * blockDim.x + threadIdx.x;
  if (idx >= N_NODES * D1) return;
  int n = idx >> 7, c = idx & 127;
  float rs = rowsum[n];
  rs = (rs == 0.f) ? 1e-12f : rs;
  float v = eluf(hraw[idx] / rs);
  size_t o = (size_t)n * D2 + head * D1 + c;
  x[o] = v;
  xh[o] = (_Float16)v;
}

__global__ void mask_scatter(const int* __restrict__ batch, float* __restrict__ mask, int nb) {
  int i = blockIdx.x * blockDim.x + threadIdx.x;
  if (i >= nb) return;
  mask[batch[i * 3 + 2]] = 1.f;
}

// final epilogue: one wave per node (8 comps/lane over 256)
__global__ void l2_final(const float* __restrict__ xlraw, const float* __restrict__ rowsum2,
                         const float* __restrict__ entup, const float* __restrict__ mask,
                         float* __restrict__ out1, float* __restrict__ outl) {
  int n = blockIdx.x * (blockDim.x >> 5) + (threadIdx.x >> 5);
  if (n >= N_NODES) return;
  int lane = threadIdx.x & 31;
  float rs = rowsum2[n];
  rs = (rs == 0.f) ? 1e-12f : rs;
  float m = mask[n];
  float a[8], b[8];
  float sa = 0.f, sb = 0.f;
  #pragma unroll
  for (int i = 0; i < 8; ++i) {
    int c = lane * 8 + i;
    float xl = xlraw[(size_t)n * D2 + c];
    float xo = eluf(xl / rs);
    float xle = eluf(xl);
    float t = entup[(size_t)n * D2 + c] + m * xo;
    a[i] = t; b[i] = xle;
    sa += t * t; sb += xle * xle;
  }
  sa = wave_sum32(sa); sb = wave_sum32(sb);
  float ia = 1.f / fmaxf(sqrtf(sa), 1e-12f);
  float ib = 1.f / fmaxf(sqrtf(sb), 1e-12f);
  #pragma unroll
  for (int i = 0; i < 8; ++i) {
    int c = lane * 8 + i;
    out1[(size_t)n * D2 + c] = a[i] * ia;
    outl[(size_t)n * D2 + c] = b[i] * ib;
  }
}

// ---------------- host launcher ----------------
static inline int cdiv(long long a, long long b) { return (int)((a + b - 1) / b); }

extern "C" void kernel_launch(void* const* d_in, const int* in_sizes, int n_in,
                              void* d_out, int out_size, void* d_ws, size_t ws_size,
                              hipStream_t stream) {
  const float* ent_in  = (const float*)d_in[0];
  const float* rel     = (const float*)d_in[1];
  const float* a1      = (const float*)d_in[2];   // [2,128,384]
  const float* a2_1    = (const float*)d_in[3];   // [2,128]
  const float* W       = (const float*)d_in[4];   // [128,256]
  const float* a_out   = (const float*)d_in[5];   // [256,768]
  const float* a2_out  = (const float*)d_in[6];   // [256]
  const float* W_ent   = (const float*)d_in[7];   // [128,256]
  const int*   batch   = (const int*)d_in[8];     // [4096,3]
  const int*   el      = (const int*)d_in[9];     // [2,300000]
  const int*   et      = (const int*)d_in[10];    // [300000]
  const int*   elnh    = (const int*)d_in[11];    // [2,30000]
  const int*   etnh    = (const int*)d_in[12];    // [30000,2]

  float* out = (float*)d_out;
  float* out1 = out;                                   // [50000,256]
  float* rel2 = out + (size_t)N_NODES * D2;            // [256,256]
  float* outl = rel2 + (size_t)N_REL * D2;             // [50000,256]

  // workspace bump allocator
  char* w = (char*)d_ws;
  auto alloc = [&](size_t bytes) -> void* {
    void* p = (void*)w;
    w += (bytes + 255) & ~(size_t)255;
    return p;
  };
  float*     ent    = (float*)alloc((size_t)N_NODES * D1 * 4);
  _Float16*  ent_h  = (_Float16*)alloc((size_t)N_NODES * D1 * 2);
  float*     pd     = (float*)alloc((size_t)N_NODES * D1 * 4);
  float*     ps     = (float*)alloc((size_t)N_NODES * D1 * 4);
  float*     qd     = (float*)alloc((size_t)N_NODES * 4);
  float*     qs     = (float*)alloc((size_t)N_NODES * 4);
  float*     hraw   = (float*)alloc((size_t)N_NODES * D1 * 4);
  float*     rsum   = (float*)alloc((size_t)N_NODES * 4);
  float*     x      = (float*)alloc((size_t)N_NODES * D2 * 4);
  _Float16*  x_h    = (_Float16*)alloc((size_t)N_NODES * D2 * 2);
  float*     pd2    = (float*)alloc((size_t)N_NODES * D2 * 4);
  float*     ps2    = (float*)alloc((size_t)N_NODES * D2 * 4);
  float*     qd2    = (float*)alloc((size_t)N_NODES * 4);
  float*     qs2    = (float*)alloc((size_t)N_NODES * 4);
  float*     xlraw  = (float*)alloc((size_t)N_NODES * D2 * 4);
  float*     rsum2  = (float*)alloc((size_t)N_NODES * 4);
  float*     entup  = (float*)alloc((size_t)N_NODES * D2 * 4);
  float*     mask   = (float*)alloc((size_t)N_NODES * 4);
  float*     prel2  = (float*)alloc((size_t)N_REL * D2 * 4);
  float*     qrel2  = (float*)alloc((size_t)N_REL * 4);
  float*     prel1  = (float*)alloc((size_t)2 * N_REL * D1 * 4);
  float*     qrel1  = (float*)alloc((size_t)2 * N_REL * 4);
  _Float16*  B1d    = (_Float16*)alloc((size_t)2 * D1 * D1 * 2);
  _Float16*  B1s    = (_Float16*)alloc((size_t)2 * D1 * D1 * 2);
  _Float16*  B2d    = (_Float16*)alloc((size_t)D2 * D2 * 2);
  _Float16*  B2s    = (_Float16*)alloc((size_t)D2 * D2 * 2);
  _Float16*  BW     = (_Float16*)alloc((size_t)D2 * D1 * 2);
  (void)ws_size; (void)in_sizes; (void)n_in; (void)out_size;

  const int BT = 256;

  // 0) zero persistent accumulators used this call
  zero_f32<<<2048, BT, 0, stream>>>(xlraw, (size_t)N_NODES * D2);
  zero_f32<<<64,   BT, 0, stream>>>(rsum2, (size_t)N_NODES);
  zero_f32<<<64,   BT, 0, stream>>>(mask,  (size_t)N_NODES);

  // 1) l2norm entities -> ent (f32) + ent_h (f16)
  l2norm128_kernel<<<cdiv((long long)N_NODES * 32, BT), BT, 0, stream>>>(ent_in, ent, ent_h);

  // 2) pack weights to f16
  for (int h = 0; h < 2; ++h) {
    pack_f16<<<cdiv(D1 * D1, BT), BT, 0, stream>>>(a1 + (size_t)h * D1 * 384, 384, 0,
                                                   B1d + (size_t)h * D1 * D1, D1, D1);
    pack_f16<<<cdiv(D1 * D1, BT), BT, 0, stream>>>(a1 + (size_t)h * D1 * 384, 384, D1,
                                                   B1s + (size_t)h * D1 * D1, D1, D1);
  }
  pack_f16<<<cdiv(D2 * D2, BT), BT, 0, stream>>>(a_out, 768, 0,   B2d, D2, D2);
  pack_f16<<<cdiv(D2 * D2, BT), BT, 0, stream>>>(a_out, 768, D2,  B2s, D2, D2);
  pack_f16_t<<<cdiv(D2 * D1, BT), BT, 0, stream>>>(W_ent, D1, D2, BW);

  // 3) out_relation_1 = rel @ W   (written straight into d_out)
  gemm_nn_f32<<<cdiv(N_REL * D2, BT), BT, 0, stream>>>(rel, W, rel2, N_REL, D2, D1);
  // prel2 = rel2 @ a_r2^T ; pr_rel1[h] = rel @ a_r1[h]^T
  gemm_nt_f32<<<cdiv(N_REL * D2, BT), BT, 0, stream>>>(rel2, a_out, 768, 512, prel2, N_REL, D2, D2);
  for (int h = 0; h < 2; ++h)
    gemm_nt_f32<<<cdiv(N_REL * D1, BT), BT, 0, stream>>>(rel, a1 + (size_t)h * D1 * 384, 384, 256,
                                                         prel1 + (size_t)h * N_REL * D1, N_REL, D1, D1);
  rowdot<<<cdiv((long long)N_REL * 32, BT), BT, 0, stream>>>(prel2, a2_out, qrel2, D2, N_REL);
  for (int h = 0; h < 2; ++h)
    rowdot<<<cdiv((long long)N_REL * 32, BT), BT, 0, stream>>>(prel1 + (size_t)h * N_REL * D1,
                                                               a2_1 + (size_t)h * D1,
                                                               qrel1 + (size_t)h * N_REL, D1, N_REL);

  // 4) layer 1, per head
  const int mtiles = N_NODES / 16;  // 3125
  for (int h = 0; h < 2; ++h) {
    int waves1 = mtiles * (D1 / 64);
    wmma_gemm_nt<<<cdiv((long long)waves1 * 32, BT), BT, 0, stream>>>(
        ent_h, D1, B1d + (size_t)h * D1 * D1, D1, pd, D1, N_NODES, D1, D1);
    wmma_gemm_nt<<<cdiv((long long)waves1 * 32, BT), BT, 0, stream>>>(
        ent_h, D1, B1s + (size_t)h * D1 * D1, D1, ps, D1, N_NODES, D1, D1);
    rowdot<<<cdiv((long long)N_NODES * 32, BT), BT, 0, stream>>>(pd, a2_1 + (size_t)h * D1, qd, D1, N_NODES);
    rowdot<<<cdiv((long long)N_NODES * 32, BT), BT, 0, stream>>>(ps, a2_1 + (size_t)h * D1, qs, D1, N_NODES);
    zero_f32<<<2048, BT, 0, stream>>>(hraw, (size_t)N_NODES * D1);
    zero_f32<<<64,   BT, 0, stream>>>(rsum, (size_t)N_NODES);
    gat_edge_d128<<<cdiv((long long)ET_CNT * 32, BT), BT, 0, stream>>>(
        el, et, elnh, etnh, pd, ps, qd, qs,
        prel1 + (size_t)h * N_REL * D1, qrel1 + (size_t)h * N_REL, hraw, rsum);
    l1_node<<<cdiv((long long)N_NODES * D1, BT), BT, 0, stream>>>(hraw, rsum, x, x_h, h);
  }

  // 5) layer 2 projections (WMMA) + logit scalars
  int waves2 = mtiles * (D2 / 64);
  wmma_gemm_nt<<<cdiv((long long)waves2 * 32, BT), BT, 0, stream>>>(
      x_h, D2, B2d, D2, pd2, D2, N_NODES, D2, D2);
  wmma_gemm_nt<<<cdiv((long long)waves2 * 32, BT), BT, 0, stream>>>(
      x_h, D2, B2s, D2, ps2, D2, N_NODES, D2, D2);
  rowdot<<<cdiv((long long)N_NODES * 32, BT), BT, 0, stream>>>(pd2, a2_out, qd2, D2, N_NODES);
  rowdot<<<cdiv((long long)N_NODES * 32, BT), BT, 0, stream>>>(ps2, a2_out, qs2, D2, N_NODES);

  // 6) entities_upgraded = ent @ W_entities (WMMA)
  wmma_gemm_nt<<<cdiv((long long)waves2 * 32, BT), BT, 0, stream>>>(
      ent_h, D1, BW, D1, entup, D2, N_NODES, D2, D1);

  // 7) layer-2 edge scatter
  gat_edge_d256<<<cdiv((long long)ET_CNT * 32, BT), BT, 0, stream>>>(
      el, et, elnh, etnh, pd2, ps2, qd2, qs2, prel2, qrel2, xlraw, rsum2);

  // 8) mask + final epilogue (out_entity_1, out_entity_l_1)
  mask_scatter<<<cdiv(4096, BT), BT, 0, stream>>>(batch, mask, 4096);
  l2_final<<<cdiv((long long)N_NODES * 32, BT), BT, 0, stream>>>(xlraw, rsum2, entup, mask, out1, outl);
}
